// DSDHLoss_73950746903182
// MI455X (gfx1250) — matrix operand order repaired and checked by reading, not compiled, not run on hardware
//
#include <hip/hip_runtime.h>
#include <hip/hip_bf16.h>
#include <math.h>
#include <stdint.h>

typedef __attribute__((ext_vector_type(16))) _Float16 v16h;
typedef __attribute__((ext_vector_type(8)))  _Float16 v8h;
typedef __attribute__((ext_vector_type(8)))  float    v8f;

#define NTRAIN 100000
#define NBIT   64
#define NCLS   100
#define NCLSP  128            // classes padded to 4 K-tiles of 32
#define NBATCH 128
#define MU_    1.0f
#define NU_    0.1f
#define LAM    0.1f           // NU/MU
#define ETAMU  55.0f          // ETA/MU
#define NCHUNK 125
#define KCHUNK 800            // 25 * 32 ; 125*800 == 100000
#define NGT    10             // upper-triangular G tiles (4x4 -> 10)
#define NTILES (NGT + 32)     // + 32 C-tiles (4x8)
#define LIKBLOCKS (NTRAIN/16) // 6250

// ---------------- WMMA fragment gathers (f16 source) ------------------------
__device__ inline v16h cat8(v8h lo, v8h hi) {
  v16h r;
#pragma unroll
  for (int h = 0; h < 8; ++h) { r[h] = lo[h]; r[h + 8] = hi[h]; }
  return r;
}

// 16-bit A-matrix 16x32 layout: lane<16: M=lane, halves0-7 -> K 0..7, halves8-15 -> K 16..23
//                               lane>=16: M=lane-16, K 8..15 / 24..31
__device__ inline v16h frag_a_h(const _Float16* X, int ld, int mr, int kb, int lane) {
  int m  = mr + (lane & 15);
  int ko = (lane & 16) ? 8 : 0;
  const _Float16* row = X + (size_t)m * ld + kb + ko;
  v8h lo = *(const v8h*)(row);       // K ko..ko+7
  v8h hi = *(const v8h*)(row + 16);  // K ko+16..ko+23
  return cat8(lo, hi);
}

// A fragment from f32 source with K bound (for y, K padded 100->128)
__device__ inline v16h frag_a_f32_guard(const float* X, int ld, int mr, int kb,
                                        int lane, int kmax) {
  int m  = mr + (lane & 15);
  int ko = (lane & 16) ? 8 : 0;
  v16h a;
#pragma unroll
  for (int h = 0; h < 16; ++h) {
    int k = kb + h + ((h & 8) ? 8 : 0) + ko;
    a[h] = (k < kmax) ? (_Float16)X[(size_t)m * ld + k] : (_Float16)0.0f;
  }
  return a;
}

// B-matrix 32x16 layout: lane<16: column N=lane, K 0..7 / 16..23 ; lane>=16: N=lane-16, K 8..15 / 24..31
// For X @ Y^T : B(K,N) = Y[nc+N][kb+K]  (contiguous row reads)
__device__ inline v16h frag_bT_h(const _Float16* Y, int ld, int nc, int kb, int lane) {
  int n = nc + (lane & 15);
  const _Float16* row = Y + (size_t)n * ld + kb + ((lane & 16) ? 8 : 0);
  v8h lo = *(const v8h*)(row);
  v8h hi = *(const v8h*)(row + 16);
  return cat8(lo, hi);
}

// For X @ Y (direct): B(K,N) = Y[kb+K][nc+N]  (column reads, coalesced across lanes)
__device__ inline v16h frag_b_h(const _Float16* Y, int ld, int kb, int nc, int lane) {
  int n  = nc + (lane & 15);
  int k0 = kb + ((lane & 16) ? 8 : 0);
  v16h b;
#pragma unroll
  for (int h = 0; h < 16; ++h) b[h] = Y[(size_t)(k0 + h + ((h & 8) ? 8 : 0)) * ld + n];
  return b;
}

// ---------------- kernels ----------------------------------------------------

// Copy U (f32), B -> f16, Ybuf -> f16 padded to 128 rows with zeros.
__global__ void k_copy(const float* __restrict__ U, const float* __restrict__ B,
                       const float* __restrict__ Y,
                       float* Uw, _Float16* Bh, _Float16* Yh) {
  size_t idx    = blockIdx.x * (size_t)blockDim.x + threadIdx.x;
  size_t stride = gridDim.x * (size_t)blockDim.x;
  const size_t nub = (size_t)NBIT * NTRAIN;
  const size_t ny  = (size_t)NCLSP * NTRAIN;
  for (size_t i = idx; i < ny; i += stride) {
    if (i < nub) { Uw[i] = U[i]; Bh[i] = (_Float16)B[i]; }
    size_t r = i / NTRAIN;
    Yh[i] = (r < NCLS) ? (_Float16)Y[i] : (_Float16)0.0f;
  }
}

// U[:, ind] = u^T ; Ybuf[:, ind] = y^T
__global__ void k_scatter(const float* __restrict__ u, const float* __restrict__ y,
                          const int* __restrict__ ind, float* Uw, _Float16* Yh) {
  int idx = blockIdx.x * blockDim.x + threadIdx.x;
  if (idx >= NBATCH * NCLSP) return;
  int s = idx / NCLSP, r = idx % NCLSP;
  int col = ind[s];
  if (r < NBIT) Uw[(size_t)r * NTRAIN + col] = u[s * NBIT + r];
  if (r < NCLS) Yh[(size_t)r * NTRAIN + col] = (_Float16)y[s * NCLS + r];
}

// Partial G = B@B^T (upper-tri tiles) and C = B@Ybuf^T (64x128) per K-chunk, WMMA f16.
__global__ void k_gc_partial(const _Float16* __restrict__ Bh,
                             const _Float16* __restrict__ Yh,
                             float* __restrict__ pG, float* __restrict__ pC) {
  int chunk = blockIdx.x;     // 0..124
  int tile  = blockIdx.y;     // 0..41
  int lane  = threadIdx.x;    // 0..31
  int kbase = chunk * KCHUNK;
  v8f acc = {};
  if (tile < NGT) {           // G upper-triangular tile
    const int TMR[NGT] = {0, 0, 0, 0, 1, 1, 1, 2, 2, 3};
    const int TNC[NGT] = {0, 1, 2, 3, 1, 2, 3, 2, 3, 3};
    int mr = TMR[tile] * 16, nc = TNC[tile] * 16;
    for (int kb = 0; kb < KCHUNK; kb += 32) {
      __builtin_prefetch(Bh + (size_t)(mr + (lane & 15)) * NTRAIN + kbase + kb + 128, 0, 3);
      v16h a = frag_a_h (Bh, NTRAIN, mr, kbase + kb, lane);
      v16h b = frag_bT_h(Bh, NTRAIN, nc, kbase + kb, lane);
      acc = __builtin_amdgcn_wmma_f32_16x16x32_f16(false, a, false, b, (short)0, acc,
                                                   false, false);
    }
    int n  = nc + (lane & 15);
    int mb = mr + ((lane & 16) ? 8 : 0);
#pragma unroll
    for (int g = 0; g < 8; ++g)
      pG[(size_t)chunk * (NBIT * NBIT) + (size_t)(mb + g) * NBIT + n] = acc[g];
  } else {                    // C tile
    int t  = tile - NGT;
    int mr = (t >> 3) * 16, nc = (t & 7) * 16;
    for (int kb = 0; kb < KCHUNK; kb += 32) {
      __builtin_prefetch(Yh + (size_t)(nc + (lane & 15)) * NTRAIN + kbase + kb + 128, 0, 3);
      v16h a = frag_a_h (Bh, NTRAIN, mr, kbase + kb, lane);
      v16h b = frag_bT_h(Yh, NTRAIN, nc, kbase + kb, lane);
      acc = __builtin_amdgcn_wmma_f32_16x16x32_f16(false, a, false, b, (short)0, acc,
                                                   false, false);
    }
    int n  = nc + (lane & 15);
    int mb = mr + ((lane & 16) ? 8 : 0);
#pragma unroll
    for (int g = 0; g < 8; ++g)
      pC[(size_t)chunk * (NBIT * NCLSP) + (size_t)(mb + g) * NCLSP + n] = acc[g];
  }
}

// Deterministic fixed-order reduction; mirrors symmetric G; adds (nu/mu) I to diag.
__global__ void k_gc_reduce(const float* __restrict__ pG, const float* __restrict__ pC,
                            float* G, float* C) {
  int idx = blockIdx.x * blockDim.x + threadIdx.x;
  if (idx < NBIT * NBIT) {
    int i = idx / NBIT, j = idx % NBIT;
    int a = (i < j) ? i : j;       // only upper-tri (a<=b) partials were written
    int b = (i < j) ? j : i;
    float s = 0.0f;
    for (int c = 0; c < NCHUNK; ++c) s += pG[(size_t)c * NBIT * NBIT + a * NBIT + b];
    G[idx] = s + ((i == j) ? LAM : 0.0f);
  } else if (idx < NBIT * NBIT + NBIT * NCLSP) {
    int k = idx - NBIT * NBIT;
    float s = 0.0f;
    for (int c = 0; c < NCHUNK; ++c) s += pC[(size_t)c * NBIT * NCLSP + k];
    C[k] = s;
  }
}

// 64x64 Cholesky, single block of 64 threads (right-looking).
__global__ void k_chol(const float* __restrict__ G, float* L) {
  __shared__ float A[NBIT * NBIT];
  int t = threadIdx.x;
  for (int j = 0; j < NBIT; ++j) A[t * NBIT + j] = G[t * NBIT + j];
  __syncthreads();
  for (int k = 0; k < NBIT; ++k) {
    if (t == k) A[k * NBIT + k] = sqrtf(A[k * NBIT + k]);
    __syncthreads();
    if (t > k) A[t * NBIT + k] /= A[k * NBIT + k];
    __syncthreads();
    if (t > k)
      for (int j = k + 1; j <= t; ++j) A[t * NBIT + j] -= A[t * NBIT + k] * A[j * NBIT + k];
    __syncthreads();
  }
  for (int j = 0; j < NBIT; ++j) L[t * NBIT + j] = (j <= t) ? A[t * NBIT + j] : 0.0f;
}

// Solve (L L^T) W = C for 100 RHS columns. One thread per RHS.
__global__ void k_solve(const float* __restrict__ L, const float* __restrict__ C,
                        float* W) {
  __shared__ float Ls[NBIT * NBIT];
  __shared__ float z[NBIT][128];
  int t = threadIdx.x;  // 128 threads, first 100 active
  for (int i = t; i < NBIT * NBIT; i += blockDim.x) Ls[i] = L[i];
  __syncthreads();
  if (t < NCLS) {
    for (int i = 0; i < NBIT; ++i) {          // forward: L z = C[:,t]
      float s = C[i * NCLSP + t];
      for (int j = 0; j < i; ++j) s -= Ls[i * NBIT + j] * z[j][t];
      z[i][t] = s / Ls[i * NBIT + i];
    }
    for (int i = NBIT - 1; i >= 0; --i) {     // backward: L^T x = z
      float s = z[i][t];
      for (int j = i + 1; j < NBIT; ++j) s -= Ls[j * NBIT + i] * z[j][t];
      z[i][t] = s / Ls[i * NBIT + i];
    }
    for (int i = 0; i < NBIT; ++i) W[i * NCLS + t] = z[i][t];
  }
}

// M = W @ W^T  (64x64, K=100)
__global__ void k_wwt(const float* __restrict__ W, float* M) {
  int idx = blockIdx.x * blockDim.x + threadIdx.x;
  if (idx >= NBIT * NBIT) return;
  int i = idx / NBIT, j = idx % NBIT;
  float s = 0.0f;
  for (int k = 0; k < NCLS; ++k) s += W[i * NCLS + k] * W[j * NCLS + k];
  M[idx] = s;
}

// Fused P = W@Ybuf + (eta/mu)U and sequential sign Gauss-Seidel, one column/lane.
// M is staged into LDS via CDNA5 async global->LDS copies (ASYNCcnt).
__global__ void __launch_bounds__(256) k_gs(const float* __restrict__ W,
                                            const float* __restrict__ M,
                                            const _Float16* __restrict__ Yh,
                                            const float* __restrict__ Uw,
                                            _Float16* __restrict__ Bh) {
  __shared__ float Wl[NBIT * NCLS];
  __shared__ float Ml[NBIT * NBIT];
  int t = threadIdx.x;
  // async fill of Ml: 4096 floats = 1024 x 16B chunks, 4 per thread
  {
    uint32_t ml_base = (uint32_t)(uintptr_t)&Ml[0];  // LDS byte offset (addr[31:0])
#pragma unroll
    for (int rep = 0; rep < 4; ++rep) {
      int cidx = rep * 256 + t;
      unsigned long long ga = (unsigned long long)(uintptr_t)(M + cidx * 4);
      uint32_t la = ml_base + cidx * 16;
      asm volatile("global_load_async_to_lds_b128 %0, %1, off"
                   :: "v"(la), "v"(ga) : "memory");
    }
  }
  for (int i = t; i < NBIT * NCLS; i += 256) Wl[i] = W[i];
  asm volatile("s_wait_asynccnt 0x0" ::: "memory");
  __syncthreads();
  int c = blockIdx.x * 256 + t;
  if (c >= NTRAIN) return;
  float P[NBIT], b[NBIT];
#pragma unroll
  for (int i = 0; i < NBIT; ++i) {
    P[i] = ETAMU * Uw[(size_t)i * NTRAIN + c];
    b[i] = (float)Bh[(size_t)i * NTRAIN + c];
  }
  for (int k = 0; k < NCLS; ++k) {
    float yk = (float)Yh[(size_t)k * NTRAIN + c];
#pragma unroll
    for (int i = 0; i < NBIT; ++i) P[i] += Wl[i * NCLS + k] * yk;
  }
#pragma unroll
  for (int i = 0; i < NBIT; ++i) {
    float acc = 0.0f;
#pragma unroll
    for (int j = 0; j < NBIT; ++j)
      if (j != i) acc += Ml[i * NBIT + j] * b[j];
    float v = P[i] - acc;
    b[i] = (v > 0.0f) ? 1.0f : ((v < 0.0f) ? -1.0f : 0.0f);
  }
#pragma unroll
  for (int i = 0; i < NBIT; ++i) Bh[(size_t)i * NTRAIN + c] = (_Float16)b[i];
}

// Likelihood over (128 x 100000): s via WMMA f16 (exact 0/1 counts), ip in f32 VALU.
__global__ void __launch_bounds__(256) k_lik(const float* __restrict__ u,
                                             const float* __restrict__ y,
                                             const float* __restrict__ Uw,
                                             const _Float16* __restrict__ Yh,
                                             double* __restrict__ part) {
  __shared__ double red[256];
  int t = threadIdx.x, wave = t >> 5, lane = t & 31;
  int nc = blockIdx.x * 16;
  int mr = wave * 16;
  v8f s = {};
#pragma unroll
  for (int kt = 0; kt < 4; ++kt) {
    v16h a = frag_a_f32_guard(y, NCLS, mr, kt * 32, lane, NCLS);
    v16h b = frag_b_h(Yh, NTRAIN, kt * 32, nc, lane);
    s = __builtin_amdgcn_wmma_f32_16x16x32_f16(false, a, false, b, (short)0, s,
                                               false, false);
  }
  int c  = nc + (lane & 15);
  int mb = mr + ((lane & 16) ? 8 : 0);
  float ip[8] = {0, 0, 0, 0, 0, 0, 0, 0};
  for (int k = 0; k < NBIT; ++k) {
    float uc = Uw[(size_t)k * NTRAIN + c];
#pragma unroll
    for (int g = 0; g < 8; ++g) ip[g] += u[(mb + g) * NBIT + k] * uc;
  }
  double ls = 0.0;
#pragma unroll
  for (int g = 0; g < 8; ++g) {
    float v  = 0.5f * ip[g];
    float sf = (s[g] > 0.0f) ? 1.0f : 0.0f;
    ls += (double)(log1pf(expf(-fabsf(v))) + fmaxf(v, 0.0f) - sf * v);
  }
  red[t] = ls;
  __syncthreads();
  for (int o = 128; o > 0; o >>= 1) {
    if (t < o) red[t] += red[t + o];
    __syncthreads();
  }
  if (t == 0) part[blockIdx.x] = red[0];
}

// Final: sum likelihood partials + cl_loss + reg_loss -> scalar.
__global__ void k_fin(const double* __restrict__ part, const float* __restrict__ W,
                      const _Float16* __restrict__ Bh, const float* __restrict__ y,
                      const int* __restrict__ ind, float* out) {
  __shared__ double r1[256], r2[256], r3[256];
  int t = threadIdx.x;
  double ls = 0.0;
  for (int i = t; i < LIKBLOCKS; i += 256) ls += part[i];
  double cs = 0.0;
  for (int idx = t; idx < NCLS * NBATCH; idx += 256) {
    int cls = idx / NBATCH, col = idx % NBATCH;
    int cc = ind[col];
    float pred = 0.0f;
    for (int i = 0; i < NBIT; ++i)
      pred += W[i * NCLS + cls] * (float)Bh[(size_t)i * NTRAIN + cc];
    float d = y[col * NCLS + cls] - pred;
    cs += (double)(d * d);
  }
  double rs = 0.0;
  for (int i = t; i < NBIT * NCLS; i += 256) { double w = W[i]; rs += w * w; }
  r1[t] = ls; r2[t] = cs; r3[t] = rs;
  __syncthreads();
  for (int o = 128; o > 0; o >>= 1) {
    if (t < o) { r1[t] += r1[t + o]; r2[t] += r2[t + o]; r3[t] += r3[t + o]; }
    __syncthreads();
  }
  if (t == 0)
    out[0] = (float)(r1[0] / ((double)NBATCH * NTRAIN)
                   + (double)MU_ * r2[0] / (NCLS * NBATCH)
                   + (double)NU_ * r3[0] / (NBIT * NCLS));
}

// ---------------- launch -----------------------------------------------------
extern "C" void kernel_launch(void* const* d_in, const int* in_sizes, int n_in,
                              void* d_out, int out_size, void* d_ws, size_t ws_size,
                              hipStream_t stream) {
  (void)in_sizes; (void)n_in; (void)out_size; (void)ws_size;
  const float* u    = (const float*)d_in[0];   // (128,64)
  const float* y    = (const float*)d_in[1];   // (128,100)
  const int*   ind  = (const int*)d_in[2];     // (128,)
  const float* U    = (const float*)d_in[3];   // (64,100000)
  const float* B    = (const float*)d_in[4];   // (64,100000)
  const float* Ybuf = (const float*)d_in[5];   // (100,100000)
  float* out = (float*)d_out;

  char* ws = (char*)d_ws;
  size_t off = 0;
  auto alloc = [&](size_t bytes) -> char* {
    char* p = ws + off;
    off = (off + bytes + 255) & ~(size_t)255;
    return p;
  };
  float*     Uw   = (float*)alloc(sizeof(float) * (size_t)NBIT * NTRAIN);
  _Float16*  Bh   = (_Float16*)alloc(sizeof(_Float16) * (size_t)NBIT * NTRAIN);
  _Float16*  Yh   = (_Float16*)alloc(sizeof(_Float16) * (size_t)NCLSP * NTRAIN);
  float*     pG   = (float*)alloc(sizeof(float) * (size_t)NCHUNK * NBIT * NBIT);
  float*     pC   = (float*)alloc(sizeof(float) * (size_t)NCHUNK * NBIT * NCLSP);
  float*     G    = (float*)alloc(sizeof(float) * NBIT * NBIT);
  float*     L    = (float*)alloc(sizeof(float) * NBIT * NBIT);
  float*     Cm   = (float*)alloc(sizeof(float) * NBIT * NCLSP);
  float*     W    = (float*)alloc(sizeof(float) * NBIT * NCLS);
  float*     Mw   = (float*)alloc(sizeof(float) * NBIT * NBIT);
  double*    part = (double*)alloc(sizeof(double) * LIKBLOCKS);

  {
    size_t ny = (size_t)NCLSP * NTRAIN;
    int blocks = (int)((ny + 255) / 256);
    k_copy<<<blocks, 256, 0, stream>>>(U, B, Ybuf, Uw, Bh, Yh);
  }
  k_scatter<<<(NBATCH * NCLSP + 255) / 256, 256, 0, stream>>>(u, y, ind, Uw, Yh);

  for (int it = 0; it < 10; ++it) {
    k_gc_partial<<<dim3(NCHUNK, NTILES), 32, 0, stream>>>(Bh, Yh, pG, pC);
    k_gc_reduce<<<(NBIT * NBIT + NBIT * NCLSP + 255) / 256, 256, 0, stream>>>(pG, pC, G, Cm);
    k_chol<<<1, 64, 0, stream>>>(G, L);
    k_solve<<<1, 128, 0, stream>>>(L, Cm, W);
    k_wwt<<<(NBIT * NBIT + 255) / 256, 256, 0, stream>>>(W, Mw);
    k_gs<<<(NTRAIN + 255) / 256, 256, 0, stream>>>(W, Mw, Yh, Uw, Bh);
  }

  k_lik<<<LIKBLOCKS, 256, 0, stream>>>(u, y, Uw, Yh, part);
  k_fin<<<1, 256, 0, stream>>>(part, W, Bh, y, ind, out);
}
